// GNN_29377576305210
// MI455X (gfx1250) — matrix operand order repaired and checked by reading, not compiled, not run on hardware
//
#include <hip/hip_runtime.h>
#include <hip/hip_bf16.h>

typedef _Float16 v16h __attribute__((ext_vector_type(16)));
typedef _Float16 v8h  __attribute__((ext_vector_type(8)));
typedef float    v8f  __attribute__((ext_vector_type(8)));

#define B_GRAPHS 1024
#define M_MAX    2000
#define P_PARTS  10

// ---------------------------------------------------------------------------
// counts -> exclusive prefix sum (B=1024, single block)
__global__ void scan_counts_kernel(const int* __restrict__ counts, int* __restrict__ cum) {
    __shared__ int s[B_GRAPHS];
    int t = threadIdx.x;
    int c = counts[t];
    s[t] = c;
    __syncthreads();
    for (int off = 1; off < B_GRAPHS; off <<= 1) {
        int v = (t >= off) ? s[t - off] : 0;
        __syncthreads();
        s[t] += v;
        __syncthreads();
    }
    cum[t] = s[t] - c;   // exclusive
}

// ---------------------------------------------------------------------------
// GCN helpers
__global__ void deg_kernel(const int* __restrict__ dst, float* __restrict__ deg, int E) {
    int e = blockIdx.x * blockDim.x + threadIdx.x;
    if (e < E) atomicAdd(&deg[dst[e]], 1.0f);
}

__global__ void dinv_kernel(float* __restrict__ deg, int N) {
    int n = blockIdx.x * blockDim.x + threadIdx.x;
    if (n < N) deg[n] = rsqrtf(1.0f + deg[n]);   // in place: deg -> dinv
}

// h = x @ W  (W: [IN][8] row-major, OUT fixed at 8)
__global__ void linear8_kernel(const float* __restrict__ x, const float* __restrict__ W,
                               float* __restrict__ h, int N, int IN) {
    int n = blockIdx.x * blockDim.x + threadIdx.x;
    if (n >= N) return;
    float xi[8];
    for (int i = 0; i < 8; ++i) xi[i] = (i < IN) ? x[(size_t)n * IN + i] : 0.0f;
    #pragma unroll
    for (int o = 0; o < 8; ++o) {
        float s = 0.0f;
        for (int i = 0; i < 8; ++i)
            if (i < IN) s += xi[i] * W[i * 8 + o];
        h[(size_t)n * 8 + o] = s;
    }
}

__global__ void msg_kernel(const int* __restrict__ src, const int* __restrict__ dst,
                           const float* __restrict__ h, const float* __restrict__ dinv,
                           float* __restrict__ msg, int E) {
    int e = blockIdx.x * blockDim.x + threadIdx.x;
    if (e >= E) return;
    int s = src[e], d = dst[e];
    float c = dinv[s] * dinv[d];
    #pragma unroll
    for (int k = 0; k < 8; ++k)
        atomicAdd(&msg[(size_t)d * 8 + k], h[(size_t)s * 8 + k] * c);
}

// out = relu(msg + h*dinv^2 + b)   (written in place over msg)
__global__ void gcn_final_kernel(const float* __restrict__ h, float* __restrict__ msg,
                                 const float* __restrict__ dinv, const float* __restrict__ bvec,
                                 int N) {
    int t = blockIdx.x * blockDim.x + threadIdx.x;
    int n = t >> 3, k = t & 7;
    if (n >= N) return;
    float di = dinv[n];
    float v = msg[(size_t)n * 8 + k] + h[(size_t)n * 8 + k] * di * di + bvec[k];
    msg[(size_t)n * 8 + k] = v > 0.0f ? v : 0.0f;
}

// scatter node features into zeroed dense batch [B][M][8] (f16)
__global__ void scatter_kernel(const float* __restrict__ h2, const int* __restrict__ batch,
                               const int* __restrict__ cum, _Float16* __restrict__ dense, int N) {
    int n = blockIdx.x * blockDim.x + threadIdx.x;
    if (n >= N) return;
    int b = batch[n];
    int pos = n - cum[b];
    _Float16* dptr = dense + ((size_t)b * M_MAX + pos) * 8;
    #pragma unroll
    for (int k = 0; k < 8; ++k) dptr[k] = (_Float16)h2[(size_t)n * 8 + k];
}

// ---------------------------------------------------------------------------
// Pack conv weights w[COUT][CIN][5] into WMMA B-fragment layout:
// wp[((nt*KC + kc)*32 + lane)*16 + i] holds im2col weight at K = kc*32 + (lane>=16)*8 + i + (i>=8)*8,
// column N = nt*16 + (lane&15); K maps to (tap, ci) as K = tap*CIN + ci. K >= KTOT -> 0.
__global__ void pack_conv_w_kernel(const float* __restrict__ w, _Float16* __restrict__ wp,
                                   int CIN, int KTOT, int KC, int NT) {
    int t = blockIdx.x * blockDim.x + threadIdx.x;
    int total = NT * KC * 32;
    if (t >= total) return;
    int lane = t & 31;
    int kc = (t >> 5) % KC;
    int nt = (t >> 5) / KC;
    int n  = nt * 16 + (lane & 15);
    int hi = lane >> 4;
    for (int i = 0; i < 16; ++i) {
        int Kg = kc * 32 + hi * 8 + i + (i >= 8 ? 8 : 0);
        _Float16 v = (_Float16)0.0f;
        if (Kg < KTOT) {
            int tap = Kg / CIN;
            int ci  = Kg - tap * CIN;
            v = (_Float16)w[((size_t)n * CIN + ci) * 5 + tap];
        }
        wp[(size_t)t * 16 + i] = v;
    }
}

// Pack fw1 [640][100] into B-fragment layout (NT=7 tiles over N, pad N>=100 with 0)
__global__ void pack_fc_w_kernel(const float* __restrict__ w, _Float16* __restrict__ wp) {
    int t = blockIdx.x * blockDim.x + threadIdx.x;
    if (t >= 7 * 20 * 32) return;
    int lane = t & 31;
    int kc = (t >> 5) % 20;
    int n  = ((t >> 5) / 20) * 16 + (lane & 15);
    int hi = lane >> 4;
    for (int i = 0; i < 16; ++i) {
        int Kg = kc * 32 + hi * 8 + i + (i >= 8 ? 8 : 0);
        _Float16 v = (_Float16)0.0f;
        if (n < 100) v = (_Float16)w[(size_t)Kg * 100 + n];
        wp[(size_t)t * 16 + i] = v;
    }
}

// ---------------------------------------------------------------------------
// Fused conv1d(k=5,'SAME') + bias + ReLU + avgpool2 via implicit-GEMM WMMA.
// in:  [B][L][CIN]  f16     out: [B][L/2][COUT] f16
// One wave -> 16 output positions x COUT channels; block = 8 waves = 128 positions.
// Key property: im2col A element (row M, reduction K) = win[(tloc+M)*CIN + K]
// (linear in K), so each A fragment is two contiguous 16B LDS loads.
template <int CIN, int COUT, int L>
__global__ __launch_bounds__(256) void conv_relu_pool_kernel(
    const _Float16* __restrict__ in, const _Float16* __restrict__ wpack,
    const float* __restrict__ bias, _Float16* __restrict__ out) {
    constexpr int KTOT = CIN * 5;
    constexpr int KC = (KTOT + 31) / 32;
    constexpr int NT = COUT / 16;
    constexpr int WIN = 136;          // 128 + halo(4) + padding so K-overreads stay in-bounds
    constexpr int CG = CIN / 8;       // v8h groups per position
    constexpr int GR = WIN * CG;      // total v8h groups
    __shared__ _Float16 win[WIN * CIN];

    const int b = blockIdx.y;
    const int blockStart = blockIdx.x * 128;

    // vectorized window staging (16B per thread-iteration), zero outside [0,L)
    for (int g = threadIdx.x; g < GR; g += 256) {
        int p = g / CG;
        int sub = g - p * CG;
        int pos = blockStart - 2 + p;
        v8h v = {};
        if (pos >= 0 && pos < L)
            v = *(const v8h*)(in + ((size_t)b * L + pos) * CIN + sub * 8);
        *(v8h*)(win + g * 8) = v;
    }
    __syncthreads();

    const int wave = threadIdx.x >> 5;
    const int lane = threadIdx.x & 31;
    const int row  = lane & 15;
    const int hi   = lane >> 4;
    const int t0   = blockStart + wave * 16;
    const int abase = (wave * 16 + row) * CIN;

    v8f acc[NT] = {};

    #pragma unroll
    for (int kc = 0; kc < KC; ++kc) {
        const int ib = abase + kc * 32 + hi * 8;
        v8h lo = *(const v8h*)(win + ib);
        v8h hh = *(const v8h*)(win + ib + 16);
        v16h a = __builtin_shufflevector(lo, hh, 0, 1, 2, 3, 4, 5, 6, 7,
                                                 8, 9, 10, 11, 12, 13, 14, 15);
        #pragma unroll
        for (int nt = 0; nt < NT; ++nt) {
            const v16h bf = *(const v16h*)(wpack + (((size_t)nt * KC + kc) * 32 + lane) * 16);
            acc[nt] = __builtin_amdgcn_wmma_f32_16x16x32_f16(false, a, false, bf,
                                                             (short)0, acc[nt], false, false);
        }
    }

    #pragma unroll
    for (int nt = 0; nt < NT; ++nt) {
        const int ch = nt * 16 + row;           // D: N = lane&15
        const float bch = bias[ch];
        #pragma unroll
        for (int p = 0; p < 4; ++p) {           // D: M = v + 8*hi -> position t0 + M
            float s0 = fmaxf(acc[nt][2 * p] + bch, 0.0f);
            float s1 = fmaxf(acc[nt][2 * p + 1] + bch, 0.0f);
            int op = (t0 >> 1) + p + hi * 4;
            if (op < (L / 2))
                out[((size_t)b * (L / 2) + op) * COUT + ch] = (_Float16)(0.5f * (s0 + s1));
        }
    }
}

// ---------------------------------------------------------------------------
// Ragged P-part mean pooling over conv3 output [B][250][64] -> flat [B][640] (c*10+p)
__global__ void pool_parts_kernel(const _Float16* __restrict__ xp, const int* __restrict__ counts,
                                  float* __restrict__ flat) {
    int b = blockIdx.x;
    int t = threadIdx.x;          // 640 threads
    int c = t & 63;
    int p = t >> 6;
    int valid = counts[b] >> 3;
    int base = valid / P_PARTS;
    int rem = valid - base * P_PARTS;
    int start = p * base + (p < rem ? p : rem);
    int size = base + (p < rem ? 1 : 0);
    float s = 0.0f;
    for (int q = 0; q < size; ++q)
        s += (float)xp[((size_t)b * 250 + start + q) * 64 + c];
    flat[(size_t)b * 640 + c * P_PARTS + p] = s / (float)size;
}

// ---------------------------------------------------------------------------
// FC1: [1024 x 640] @ [640 x 100(->112)] via WMMA; relu epilogue.
// One wave per 16-batch tile; kc outer, 7 N-tiles inner with persistent accumulators.
__global__ __launch_bounds__(256) void fc1_wmma_kernel(const float* __restrict__ flat,
                                                       const _Float16* __restrict__ wpack,
                                                       const float* __restrict__ fb1,
                                                       float* __restrict__ out) {
    const int gw = (blockIdx.x * 256 + threadIdx.x) >> 5;   // 0..63
    const int lane = threadIdx.x & 31;
    const int row = lane & 15;
    const int hi = lane >> 4;
    const int b0 = gw * 16;
    const int bb = b0 + row;

    v8f acc[7] = {};

    for (int kc = 0; kc < 20; ++kc) {
        const float* fp = flat + (size_t)bb * 640 + kc * 32 + hi * 8;
        v16h a;
        #pragma unroll
        for (int i = 0; i < 8; ++i) a[i] = (_Float16)fp[i];
        #pragma unroll
        for (int i = 0; i < 8; ++i) a[8 + i] = (_Float16)fp[16 + i];
        #pragma unroll
        for (int nt = 0; nt < 7; ++nt) {
            const v16h bf = *(const v16h*)(wpack + (((size_t)nt * 20 + kc) * 32 + lane) * 16);
            acc[nt] = __builtin_amdgcn_wmma_f32_16x16x32_f16(false, a, false, bf,
                                                             (short)0, acc[nt], false, false);
        }
    }

    #pragma unroll
    for (int nt = 0; nt < 7; ++nt) {
        int n = nt * 16 + row;
        if (n < 100) {
            float bias = fb1[n];
            #pragma unroll
            for (int v = 0; v < 8; ++v) {
                int bo = b0 + v + 8 * hi;               // D: M = v + 8*hi
                float r = acc[nt][v] + bias;
                out[(size_t)bo * 100 + n] = r > 0.0f ? r : 0.0f;
            }
        }
    }
}

// FC2: [B,100] @ [100,2] + bias
__global__ void fc2_kernel(const float* __restrict__ a, const float* __restrict__ w,
                           const float* __restrict__ bias, float* __restrict__ out) {
    int t = blockIdx.x * blockDim.x + threadIdx.x;
    if (t >= B_GRAPHS * 2) return;
    int b = t >> 1, j = t & 1;
    float s = bias[j];
    for (int k = 0; k < 100; ++k) s += a[(size_t)b * 100 + k] * w[k * 2 + j];
    out[(size_t)b * 2 + j] = s;
}

// ---------------------------------------------------------------------------
extern "C" void kernel_launch(void* const* d_in, const int* in_sizes, int n_in,
                              void* d_out, int out_size, void* d_ws, size_t ws_size,
                              hipStream_t stream) {
    const float* x      = (const float*)d_in[0];
    const int*   src    = (const int*)d_in[1];
    const int*   dst    = (const int*)d_in[2];
    const int*   batch  = (const int*)d_in[3];
    const int*   counts = (const int*)d_in[4];
    const float* W1  = (const float*)d_in[5];
    const float* b1  = (const float*)d_in[6];
    const float* W2  = (const float*)d_in[7];
    const float* b2  = (const float*)d_in[8];
    const float* cw1 = (const float*)d_in[9];
    const float* cb1 = (const float*)d_in[10];
    const float* cw2 = (const float*)d_in[11];
    const float* cb2 = (const float*)d_in[12];
    const float* cw3 = (const float*)d_in[13];
    const float* cb3 = (const float*)d_in[14];
    const float* fw1 = (const float*)d_in[15];
    const float* fb1 = (const float*)d_in[16];
    const float* fw2 = (const float*)d_in[17];
    const float* fb2 = (const float*)d_in[18];
    float* outp = (float*)d_out;

    const int N = in_sizes[0] / 4;   // nodes
    const int E = in_sizes[1];       // edges

    // ---- carve workspace (256B aligned slices) ----
    size_t off = 0;
    auto take = [&](size_t bytes) -> void* {
        void* p = (char*)d_ws + off;
        off += (bytes + 255) & ~(size_t)255;
        return p;
    };
    int*       cum   = (int*)take((size_t)B_GRAPHS * 4);
    float*     dinv  = (float*)take((size_t)N * 4);              // deg -> dinv in place
    float*     bufA  = (float*)take((size_t)N * 8 * 4);
    float*     bufB  = (float*)take((size_t)N * 8 * 4);
    float*     bufC  = (float*)take((size_t)N * 8 * 4);
    _Float16*  dense = (_Float16*)take((size_t)B_GRAPHS * M_MAX * 8 * 2);
    _Float16*  c1out = (_Float16*)take((size_t)B_GRAPHS * 1000 * 16 * 2);
    _Float16*  c2out = (_Float16*)take((size_t)B_GRAPHS * 500 * 32 * 2);
    _Float16*  c3out = (_Float16*)take((size_t)B_GRAPHS * 250 * 64 * 2);
    _Float16*  wp1   = (_Float16*)take((size_t)1 * 2 * 32 * 16 * 2);
    _Float16*  wp2   = (_Float16*)take((size_t)2 * 3 * 32 * 16 * 2);
    _Float16*  wp3   = (_Float16*)take((size_t)4 * 5 * 32 * 16 * 2);
    _Float16*  wpf   = (_Float16*)take((size_t)7 * 20 * 32 * 16 * 2);
    float*     flat  = (float*)take((size_t)B_GRAPHS * 640 * 4);
    float*     fc1o  = (float*)take((size_t)B_GRAPHS * 100 * 4);
    (void)ws_size; (void)n_in; (void)out_size;

    const int TB = 256;
    auto blocks = [&](long long n) { return (int)((n + TB - 1) / TB); };

    // prefix offsets of graphs
    scan_counts_kernel<<<1, B_GRAPHS, 0, stream>>>(counts, cum);

    // degree + rsqrt normalization
    hipMemsetAsync(dinv, 0, (size_t)N * 4, stream);
    deg_kernel<<<blocks(E), TB, 0, stream>>>(dst, dinv, E);
    dinv_kernel<<<blocks(N), TB, 0, stream>>>(dinv, N);

    // GCN layer 1: h=x@W1 -> bufA; msg -> bufB; relu(msg + h*dinv^2 + b1) in bufB
    linear8_kernel<<<blocks(N), TB, 0, stream>>>(x, W1, bufA, N, 4);
    hipMemsetAsync(bufB, 0, (size_t)N * 8 * 4, stream);
    msg_kernel<<<blocks(E), TB, 0, stream>>>(src, dst, bufA, dinv, bufB, E);
    gcn_final_kernel<<<blocks((long long)N * 8), TB, 0, stream>>>(bufA, bufB, dinv, b1, N);

    // GCN layer 2: bufB -> bufA(h) -> bufC
    linear8_kernel<<<blocks(N), TB, 0, stream>>>(bufB, W2, bufA, N, 8);
    hipMemsetAsync(bufC, 0, (size_t)N * 8 * 4, stream);
    msg_kernel<<<blocks(E), TB, 0, stream>>>(src, dst, bufA, dinv, bufC, E);
    gcn_final_kernel<<<blocks((long long)N * 8), TB, 0, stream>>>(bufA, bufC, dinv, b2, N);

    // to_dense_batch (f16, zero padded)
    hipMemsetAsync(dense, 0, (size_t)B_GRAPHS * M_MAX * 8 * 2, stream);
    scatter_kernel<<<blocks(N), TB, 0, stream>>>(bufC, batch, cum, dense, N);

    // pack weights into WMMA B-fragment layout
    pack_conv_w_kernel<<<blocks(1 * 2 * 32), TB, 0, stream>>>(cw1, wp1, 8, 40, 2, 1);
    pack_conv_w_kernel<<<blocks(2 * 3 * 32), TB, 0, stream>>>(cw2, wp2, 16, 80, 3, 2);
    pack_conv_w_kernel<<<blocks(4 * 5 * 32), TB, 0, stream>>>(cw3, wp3, 32, 160, 5, 4);
    pack_fc_w_kernel<<<blocks(7 * 20 * 32), TB, 0, stream>>>(fw1, wpf);

    // fused conv+relu+pool stack (implicit GEMM, WMMA f16->f32)
    conv_relu_pool_kernel<8, 16, 2000><<<dim3(16, B_GRAPHS), TB, 0, stream>>>(dense, wp1, cb1, c1out);
    conv_relu_pool_kernel<16, 32, 1000><<<dim3(8, B_GRAPHS), TB, 0, stream>>>(c1out, wp2, cb2, c2out);
    conv_relu_pool_kernel<32, 64, 500><<<dim3(4, B_GRAPHS), TB, 0, stream>>>(c2out, wp3, cb3, c3out);

    // ragged part-mean pooling -> [B,640]
    pool_parts_kernel<<<B_GRAPHS, 640, 0, stream>>>(c3out, counts, flat);

    // MLP head
    fc1_wmma_kernel<<<8, TB, 0, stream>>>(flat, wpf, fb1, fc1o);
    fc2_kernel<<<blocks(B_GRAPHS * 2), TB, 0, stream>>>(fc1o, fw2, fb2, outp);
}
